// FusedAttention_74131135529907
// MI455X (gfx1250) — matrix operand order repaired and checked by reading, not compiled, not run on hardware
//
#include <hip/hip_runtime.h>
#include <hip/hip_bf16.h>

// ---------------------------------------------------------------------------
// Fused multi-head attention for MI455X (gfx1250, wave32, WMMA bf16)
//   d_model=1024, heads=16, d_k=64, B=2, S=2048
// Pipeline: fp32->bf16 convert, QKV GEMM (WMMA, 32x64 tile/wave),
// flash-attention (WMMA + online softmax, scores never hit HBM),
// output GEMM (WMMA, 32x64 tile/wave, fp32 out).
// ---------------------------------------------------------------------------

typedef __bf16 bf16_t;
typedef __attribute__((ext_vector_type(16))) __bf16 v16bf;
typedef __attribute__((ext_vector_type(8)))  __bf16 v8bf;
typedef __attribute__((ext_vector_type(8)))  float  v8f;

#define D_MODEL 1024
#define NHEADS  16
#define DK      64
#define SEQ     2048
#define BATCH   2
#define NROWS   (BATCH * SEQ)        // 4096 rows of x / ctx
#define SCALE   0.125f               // 1/sqrt(64)

// ---------------------------------------------------------------------------
// Fragment builders, per CDNA5 ISA 7.12.2 (wave32 layouts).
// ---------------------------------------------------------------------------

// A-matrix 16x32 bf16 fragment. Source: row-major [M>=16, stride] at (row0=0,
// col k0). Lanes 0-15: row=lane, halves 0-7 = K k0+0..7, halves 8-15 = K
// k0+16..23. Lanes 16-31: row=lane-16, K k0+8..15 and k0+24..31.
static __device__ inline v16bf make_afrag(const bf16_t* __restrict__ base,
                                          int stride, int lane, int k0) {
  const int row  = lane & 15;
  const int ksel = (lane & 16) ? 8 : 0;
  const bf16_t* p = base + row * stride + k0 + ksel;
  v8bf lo = *(const v8bf*)(p);
  v8bf hi = *(const v8bf*)(p + 16);
  v16bf a;
#pragma unroll
  for (int i = 0; i < 8; ++i) { a[i] = lo[i]; a[8 + i] = hi[i]; }
  return a;
}

// B-matrix 32x16 bf16 fragment where the SOURCE is row-major [N, K] (i.e.
// B = src^T, B[k][n] = src[n][k]); covers N=0..15 of `base`, K=k0..k0+31.
// Lanes 0-15: N=lane, K=k0+0..7 and k0+16..23. Lanes 16-31: N=lane-16,
// K=k0+8..15 and k0+24..31. Two 16-byte loads per lane, no transpose needed.
static __device__ inline v16bf make_bfrag_nk(const bf16_t* __restrict__ base,
                                             int stride, int lane, int k0) {
  const int n  = lane & 15;
  const int kk = k0 + ((lane & 16) ? 8 : 0);
  const bf16_t* p = base + n * stride + kk;
  v8bf lo = *(const v8bf*)(p);
  v8bf hi = *(const v8bf*)(p + 16);
  v16bf b;
#pragma unroll
  for (int i = 0; i < 8; ++i) { b[i] = lo[i]; b[8 + i] = hi[i]; }
  return b;
}

static __device__ inline v8f wmma_bf16(v16bf a, v16bf b, v8f c) {
  // D(16x16 f32) = A(16x32 bf16) x B(32x16 bf16) + C
  return __builtin_amdgcn_wmma_f32_16x16x32_bf16(
      /*neg_a=*/false, a, /*neg_b=*/false, b,
      /*c_mod=*/(short)0, c, /*reuse_a=*/false, /*reuse_b=*/false);
}

// ---------------------------------------------------------------------------
// fp32 -> bf16 conversion
// ---------------------------------------------------------------------------
__global__ void cvt_bf16(const float* __restrict__ src, bf16_t* __restrict__ dst, int n) {
  int i = blockIdx.x * blockDim.x + threadIdx.x;
  if (i < n) dst[i] = (bf16_t)src[i];
}

// ---------------------------------------------------------------------------
// QKV projection: C[4096,3072] = X[4096,1024] * Wqkv^T (W is [3072,1024]).
// Each wave computes a 32x64 tile (2 A-frags share 4 B-frags -> 1.5 b128
// loads per WMMA); results scattered into Q/K/V stored as [b, h, s, dk] bf16.
// ---------------------------------------------------------------------------
__global__ __launch_bounds__(256) void qkv_gemm(
    const bf16_t* __restrict__ X, const bf16_t* __restrict__ W,
    bf16_t* __restrict__ Qo, bf16_t* __restrict__ Ko, bf16_t* __restrict__ Vo) {
  const int lane = threadIdx.x & 31;
  const int wid  = blockIdx.x * (blockDim.x >> 5) + (threadIdx.x >> 5);
  const int NT   = (3 * D_MODEL) / 64;         // 48 column groups of 64
  const int mt   = wid / NT;
  const int ng   = wid % NT;
  const int m0   = mt * 32;
  const int n0   = ng * 64;

  v8f acc[2][4] = {};
  const bf16_t* arow0 = X + (size_t)m0 * D_MODEL;
  const bf16_t* arow1 = arow0 + (size_t)16 * D_MODEL;
  for (int k0 = 0; k0 < D_MODEL; k0 += 32) {
    v16bf a0 = make_afrag(arow0, D_MODEL, lane, k0);
    v16bf a1 = make_afrag(arow1, D_MODEL, lane, k0);
#pragma unroll
    for (int t = 0; t < 4; ++t) {
      v16bf b = make_bfrag_nk(W + (size_t)(n0 + t * 16) * D_MODEL, D_MODEL, lane, k0);
      acc[0][t] = wmma_bf16(a0, b, acc[0][t]);
      acc[1][t] = wmma_bf16(a1, b, acc[1][t]);
    }
  }

  // C layout: VGPR r -> row r + 8*(lane>=16); col = lane&15 within the n-tile.
  const int col = lane & 15;
#pragma unroll
  for (int rg = 0; rg < 2; ++rg) {
    const int rbase = m0 + rg * 16 + ((lane & 16) ? 8 : 0);
#pragma unroll
    for (int t = 0; t < 4; ++t) {
      const int e    = n0 + t * 16 + col;        // 0..3071
      const int part = e >> 10;                  // 0=Q 1=K 2=V
      const int eh   = e & (D_MODEL - 1);
      const int h    = eh >> 6;
      const int dk   = eh & (DK - 1);
      bf16_t* dst = (part == 0) ? Qo : (part == 1) ? Ko : Vo;
#pragma unroll
      for (int r = 0; r < 8; ++r) {
        const int m = rbase + r;
        const int b = m >> 11;                   // /SEQ
        const int s = m & (SEQ - 1);
        dst[(((size_t)(b * NHEADS + h) * SEQ + s) << 6) + dk] = (bf16_t)acc[rg][t][r];
      }
    }
  }
}

// ---------------------------------------------------------------------------
// Flash attention. Block = 4 waves (128 threads) handles (b*h, 64 queries).
// Waves cooperatively stage 32-key K and V^T chunks into LDS; each wave owns
// 16 queries: QK^T (2 WMMA x 2 n-tiles), online softmax, P via LDS relayout,
// PV (4 WMMA). Output ctx written bf16 in [b, s, d_model] layout.
// ---------------------------------------------------------------------------
__global__ __launch_bounds__(128) void attn_kernel(
    const bf16_t* __restrict__ Qg, const bf16_t* __restrict__ Kg,
    const bf16_t* __restrict__ Vg, bf16_t* __restrict__ CTX) {
  __shared__ bf16_t Ksh[32 * DK];        // [key][dk]
  __shared__ bf16_t Vsh[DK * 32];        // transposed: [dk][key]
  __shared__ bf16_t Psh[4][16 * 32];     // per-wave P staging

  const int lane = threadIdx.x & 31;
  const int wv   = threadIdx.x >> 5;                 // 0..3
  const int bh   = blockIdx.x >> 5;                  // 0..31  (b*16+h)
  const int qb   = blockIdx.x & 31;                  // 0..31  query block
  const int q0   = qb * 64 + wv * 16;                // first query of this wave

  const bf16_t* Qh = Qg + (size_t)bh * SEQ * DK;
  const bf16_t* Kh = Kg + (size_t)bh * SEQ * DK;
  const bf16_t* Vh = Vg + (size_t)bh * SEQ * DK;

  // Q fragments for this wave's 16 queries (K-dim = 64 -> two 16x32 frags).
  const v16bf qa0 = make_afrag(Qh + (size_t)q0 * DK, DK, lane, 0);
  const v16bf qa1 = make_afrag(Qh + (size_t)q0 * DK, DK, lane, 32);

  float mi[8], li[8];
  v8f   acc[4] = {};
#pragma unroll
  for (int r = 0; r < 8; ++r) { mi[r] = -1e30f; li[r] = 0.0f; }

  const int col = lane & 15;               // C-matrix column for this lane

  for (int j = 0; j < SEQ; j += 32) {
    __syncthreads();                        // previous chunk fully consumed

    // --- cooperative stage: K (flat copy) and V (transposed) ---
    {
      const int t = threadIdx.x;            // 0..127
      const v8bf* srck = (const v8bf*)(Kh + (size_t)j * DK);
      v8bf* dstk = (v8bf*)Ksh;
      dstk[t] = srck[t];
      dstk[t + 128] = srck[t + 128];

      const v8bf* srcv = (const v8bf*)(Vh + (size_t)j * DK);
#pragma unroll
      for (int it = 0; it < 2; ++it) {
        const int vi = t + it * 128;        // 0..255 vectors of 8
        const int s  = vi >> 3;             // key in chunk
        const int d0 = (vi & 7) * 8;        // first dk of this vector
        v8bf vv = srcv[vi];
#pragma unroll
        for (int i = 0; i < 8; ++i) Vsh[(d0 + i) * 32 + s] = vv[i];
      }
    }
    __syncthreads();

    // --- scores: S[16x32] = Q[16x64] * K_chunk^T, scaled ---
    v8f s0 = {}, s1 = {};
    {
      v16bf b00 = make_bfrag_nk(Ksh, DK, lane, 0);             // keys 0-15
      v16bf b01 = make_bfrag_nk(Ksh, DK, lane, 32);
      s0 = wmma_bf16(qa0, b00, s0);
      s0 = wmma_bf16(qa1, b01, s0);
      v16bf b10 = make_bfrag_nk(Ksh + 16 * DK, DK, lane, 0);   // keys 16-31
      v16bf b11 = make_bfrag_nk(Ksh + 16 * DK, DK, lane, 32);
      s1 = wmma_bf16(qa0, b10, s1);
      s1 = wmma_bf16(qa1, b11, s1);
    }

    // --- online softmax (rows striped: row = r + 8*(lane>=16), col = lane&15;
    //     lanes 0-15 and 16-31 hold disjoint rows, so xor-reduce over 16) ---
    float p0[8], p1[8], alpha[8];
#pragma unroll
    for (int r = 0; r < 8; ++r) {
      p0[r] = s0[r] * SCALE;
      p1[r] = s1[r] * SCALE;
      float v = fmaxf(p0[r], p1[r]);
      v = fmaxf(v, __shfl_xor(v, 8, 32));
      v = fmaxf(v, __shfl_xor(v, 4, 32));
      v = fmaxf(v, __shfl_xor(v, 2, 32));
      v = fmaxf(v, __shfl_xor(v, 1, 32));
      const float m_new = fmaxf(mi[r], v);
      alpha[r] = __expf(mi[r] - m_new);
      p0[r] = __expf(p0[r] - m_new);
      p1[r] = __expf(p1[r] - m_new);
      float rs = p0[r] + p1[r];
      rs += __shfl_xor(rs, 8, 32);
      rs += __shfl_xor(rs, 4, 32);
      rs += __shfl_xor(rs, 2, 32);
      rs += __shfl_xor(rs, 1, 32);
      li[r] = li[r] * alpha[r] + rs;
      mi[r] = m_new;
    }
#pragma unroll
    for (int t = 0; t < 4; ++t)
#pragma unroll
      for (int r = 0; r < 8; ++r) acc[t][r] *= alpha[r];

    // --- relayout P through per-wave LDS into an A-fragment ---
    {
      bf16_t* pw = &Psh[wv][0];
      const int rowoff = (lane & 16) ? 8 : 0;
#pragma unroll
      for (int r = 0; r < 8; ++r) {
        pw[(r + rowoff) * 32 + col]      = (bf16_t)p0[r];
        pw[(r + rowoff) * 32 + col + 16] = (bf16_t)p1[r];
      }
      asm volatile("s_wait_dscnt 0" ::: "memory");   // wave-local RAW on LDS
      v16bf pa = make_afrag(pw, 32, lane, 0);

      // --- PV: acc[16x64] += P[16x32] * V_chunk[32x64] (V^T in LDS) ---
#pragma unroll
      for (int t = 0; t < 4; ++t) {
        v16bf vb = make_bfrag_nk(Vsh + (t * 16) * 32, 32, lane, 0);
        acc[t] = wmma_bf16(pa, vb, acc[t]);
      }
    }
  }

  // --- normalize + store ctx as bf16 [b, s, d_model] ---
  const int b = bh >> 4;
  const int h = bh & (NHEADS - 1);
  const int rowoff = (lane & 16) ? 8 : 0;
  float inv[8];
#pragma unroll
  for (int r = 0; r < 8; ++r) inv[r] = 1.0f / li[r];
#pragma unroll
  for (int t = 0; t < 4; ++t) {
#pragma unroll
    for (int r = 0; r < 8; ++r) {
      const int m = q0 + rowoff + r;               // query index
      const int d = t * 16 + col;                  // 0..63
      CTX[((size_t)(b * SEQ + m)) * D_MODEL + h * DK + d] =
          (bf16_t)(acc[t][r] * inv[r]);
    }
  }
}

// ---------------------------------------------------------------------------
// Output projection: OUT[4096,1024] (fp32) = CTX[4096,1024] * Wout^T.
// 32x64 tile per wave, same fragment reuse scheme as qkv_gemm.
// ---------------------------------------------------------------------------
__global__ __launch_bounds__(256) void out_gemm(
    const bf16_t* __restrict__ CTX, const bf16_t* __restrict__ W,
    float* __restrict__ OUT) {
  const int lane = threadIdx.x & 31;
  const int wid  = blockIdx.x * (blockDim.x >> 5) + (threadIdx.x >> 5);
  const int NT   = D_MODEL / 64;               // 16 column groups
  const int mt   = wid / NT;
  const int ng   = wid % NT;
  const int m0   = mt * 32;
  const int n0   = ng * 64;

  v8f acc[2][4] = {};
  const bf16_t* arow0 = CTX + (size_t)m0 * D_MODEL;
  const bf16_t* arow1 = arow0 + (size_t)16 * D_MODEL;
  for (int k0 = 0; k0 < D_MODEL; k0 += 32) {
    v16bf a0 = make_afrag(arow0, D_MODEL, lane, k0);
    v16bf a1 = make_afrag(arow1, D_MODEL, lane, k0);
#pragma unroll
    for (int t = 0; t < 4; ++t) {
      v16bf b = make_bfrag_nk(W + (size_t)(n0 + t * 16) * D_MODEL, D_MODEL, lane, k0);
      acc[0][t] = wmma_bf16(a0, b, acc[0][t]);
      acc[1][t] = wmma_bf16(a1, b, acc[1][t]);
    }
  }

  const int col = lane & 15;
#pragma unroll
  for (int rg = 0; rg < 2; ++rg) {
    const int rbase = m0 + rg * 16 + ((lane & 16) ? 8 : 0);
#pragma unroll
    for (int t = 0; t < 4; ++t) {
#pragma unroll
      for (int r = 0; r < 8; ++r) {
        OUT[(size_t)(rbase + r) * D_MODEL + n0 + t * 16 + col] = acc[rg][t][r];
      }
    }
  }
}

// ---------------------------------------------------------------------------
// Host-side launcher
// ---------------------------------------------------------------------------
extern "C" void kernel_launch(void* const* d_in, const int* in_sizes, int n_in,
                              void* d_out, int out_size, void* d_ws, size_t ws_size,
                              hipStream_t stream) {
  (void)in_sizes; (void)n_in; (void)out_size; (void)ws_size;
  const float* x     = (const float*)d_in[0];   // [2,2048,1024]
  const float* w_qkv = (const float*)d_in[1];   // [3072,1024]
  const float* w_out = (const float*)d_in[2];   // [1024,1024]
  float* out = (float*)d_out;                   // [2,2048,1024]

  // Workspace carve-up (bf16 elements), total ~48 MB.
  bf16_t* xb    = (bf16_t*)d_ws;                         // 4096*1024
  bf16_t* wqkvb = xb    + (size_t)NROWS * D_MODEL;       // 3072*1024
  bf16_t* woutb = wqkvb + (size_t)3 * D_MODEL * D_MODEL; // 1024*1024
  bf16_t* q     = woutb + (size_t)D_MODEL * D_MODEL;     // 32*2048*64
  bf16_t* k     = q     + (size_t)BATCH * NHEADS * SEQ * DK;
  bf16_t* v     = k     + (size_t)BATCH * NHEADS * SEQ * DK;
  bf16_t* ctx   = v     + (size_t)BATCH * NHEADS * SEQ * DK;

  const int nx    = NROWS * D_MODEL;
  const int nwqkv = 3 * D_MODEL * D_MODEL;
  const int nwout = D_MODEL * D_MODEL;
  cvt_bf16<<<(nx    + 255) / 256, 256, 0, stream>>>(x,     xb,    nx);
  cvt_bf16<<<(nwqkv + 255) / 256, 256, 0, stream>>>(w_qkv, wqkvb, nwqkv);
  cvt_bf16<<<(nwout + 255) / 256, 256, 0, stream>>>(w_out, woutb, nwout);

  // QKV GEMM: (4096/32) * (3072/64) = 6144 waves, 8 waves per 256-thr block.
  qkv_gemm<<<768, 256, 0, stream>>>(xb, wqkvb, q, k, v);

  // Attention: 32 (b*h) * 32 query-blocks = 1024 blocks of 128 threads.
  attn_kernel<<<1024, 128, 0, stream>>>(q, k, v, ctx);

  // Output GEMM: (4096/32) * (1024/64) = 2048 waves -> 256 blocks.
  out_gemm<<<256, 256, 0, stream>>>(ctx, woutb, out);
}
